// VideoContrastiveLoss_32564442038466
// MI455X (gfx1250) — matrix operand order repaired and checked by reading, not compiled, not run on hardware
//
#include <hip/hip_runtime.h>
#include <hip/hip_bf16.h>

typedef __attribute__((ext_vector_type(16))) _Float16 v16h;
typedef __attribute__((ext_vector_type(8)))  _Float16 v8h;
typedef __attribute__((ext_vector_type(4)))  _Float16 v4h;
typedef __attribute__((ext_vector_type(8)))  float    v8f;

#define N_ROWS 8192
#define DIM    1024
#define HALF_N 4096
#define INV_T  14.2857142857142857f   /* 1 / 0.07 ; also the logit upper bound */

// ds_swizzle_b32 group-of-32 xor butterflies (single DS op, no LDS memory,
// no lane-index VALU; avoids __shfl_xor's ds_bpermute + clamping overhead).
// offset encoding: [14:10]=xor_mask, [9:5]=or_mask, [4:0]=and_mask.
#define SWZ_ADD(v, pat)                                                        \
    ((v) + __int_as_float(__builtin_amdgcn_ds_swizzle(__float_as_int(v), (pat))))
#define SWAPX1  0x041f
#define SWAPX2  0x081f
#define SWAPX4  0x101f
#define SWAPX8  0x201f
#define SWAPX16 0x401f

__device__ __forceinline__ float wave32_sum(float v) {
    v = SWZ_ADD(v, SWAPX1);
    v = SWZ_ADD(v, SWAPX2);
    v = SWZ_ADD(v, SWAPX4);
    v = SWZ_ADD(v, SWAPX8);
    v = SWZ_ADD(v, SWAPX16);
    return v;
}

// ---------------------------------------------------------------------------
// Kernel 1: row L2-normalize fp32 -> f16, zero rowsum accumulator and output.
// One block (256 threads) per row; each thread handles 4 contiguous floats.
// ---------------------------------------------------------------------------
__global__ __launch_bounds__(256)
void vcl_normalize(const float* __restrict__ f, _Float16* __restrict__ fn,
                   float* __restrict__ rowsum, float* __restrict__ out) {
    const int row = blockIdx.x;
    const int tid = threadIdx.x;

    const float4 x = ((const float4*)(f + (size_t)row * DIM))[tid];
    float ss = x.x * x.x + x.y * x.y + x.z * x.z + x.w * x.w;
    ss = wave32_sum(ss);

    __shared__ float ws[8];
    if ((tid & 31) == 0) ws[tid >> 5] = ss;
    __syncthreads();
    float tot = 0.f;
    #pragma unroll
    for (int w = 0; w < 8; ++w) tot += ws[w];

    const float inv = 1.0f / fmaxf(sqrtf(tot), 1e-8f);

    v4h h;
    h[0] = (_Float16)(x.x * inv);
    h[1] = (_Float16)(x.y * inv);
    h[2] = (_Float16)(x.z * inv);
    h[3] = (_Float16)(x.w * inv);
    ((v4h*)(fn + (size_t)row * DIM))[tid] = h;

    if (tid == 0) rowsum[row] = 0.0f;
    if (row == 0 && tid == 0) out[0] = 0.0f;
}

// ---------------------------------------------------------------------------
// Kernel 2: fused WMMA GEMM (cos = fn @ fn^T) + online exp-sum epilogue.
// Block tile 128x128, 8 waves arranged 4(M) x 2(N); each wave computes a
// 32x64 subtile with v_wmma_f32_16x16x32_f16 (2 M-frags x 4 N-frags).
// Fragments loaded straight from global (entire f16 matrix is L2-resident).
//
// A frag (16x32, MxK), contiguous k:   lane(l) hi=l>>4, m=l&15
//   halves a[0..7]  = fn[M0+m][kb + hi*16 + 0..7]
//   halves a[8..15] = fn[M0+m][kb + hi*16 + 8..15]
// B frag (32x16, KxN), B[k][n] = fn[N0+n][kb+k]:  lane(l) hi=l>>4, n=l&15
//   halves b[0..15] = fn[N0+n][kb + hi*16 + 0..15]   (32 contiguous bytes)
// C/D frag: vgpr r, lanes 0-15 -> M=r, lanes 16-31 -> M=r+8; N = l&15.
//
// Epilogue: cosine <= 1, so exp((cos-1)/T) <= 1: fixed-shift logsumexp is
// stable with no row-max pass.  Pair logit handled in the finalize kernel,
// so the epilogue is branch-free except for the row-sum atomic lane.
// ---------------------------------------------------------------------------
__global__ __launch_bounds__(256)
void vcl_gemm(const _Float16* __restrict__ fn, float* __restrict__ rowsum) {
    const int lane  = threadIdx.x & 31;
    const int wave  = threadIdx.x >> 5;
    const int hi    = lane >> 4;      // half-wave select
    const int lx    = lane & 15;
    const int mwave = wave >> 1;      // 0..3
    const int nwave = wave & 1;       // 0..1
    const int mbase = blockIdx.y * 128 + mwave * 32;
    const int nbase = blockIdx.x * 128 + nwave * 64;

    v8f acc[2][4];
    #pragma unroll
    for (int mi = 0; mi < 2; ++mi)
        #pragma unroll
        for (int ni = 0; ni < 4; ++ni) acc[mi][ni] = v8f{};

    for (int kb = 0; kb < DIM; kb += 32) {
        v16h a[2], b[4];
        #pragma unroll
        for (int mi = 0; mi < 2; ++mi) {
            const _Float16* p = fn + (size_t)(mbase + mi * 16 + lx) * DIM + kb;
            const v8h c0 = *(const v8h*)(p + hi * 16);       // K = hi*16 + 0..7
            const v8h c1 = *(const v8h*)(p + hi * 16 + 8);   // K = hi*16 + 8..15
            #pragma unroll
            for (int e = 0; e < 8; ++e) { a[mi][e] = c0[e]; a[mi][8 + e] = c1[e]; }
        }
        #pragma unroll
        for (int ni = 0; ni < 4; ++ni) {
            const _Float16* p = fn + (size_t)(nbase + ni * 16 + lx) * DIM + kb + hi * 16;
            b[ni] = *(const v16h*)p;                          // 32 contiguous bytes
        }
        #pragma unroll
        for (int mi = 0; mi < 2; ++mi)
            #pragma unroll
            for (int ni = 0; ni < 4; ++ni)
                acc[mi][ni] = __builtin_amdgcn_wmma_f32_16x16x32_f16(
                    /*neg_a=*/false, a[mi], /*neg_b=*/false, b[ni],
                    /*c_mod=*/(short)0, acc[mi][ni],
                    /*reuse_a=*/false, /*reuse_b=*/false);
    }

    #pragma unroll
    for (int mi = 0; mi < 2; ++mi) {
        float esum[8];
        #pragma unroll
        for (int r = 0; r < 8; ++r) esum[r] = 0.0f;

        #pragma unroll
        for (int ni = 0; ni < 4; ++ni) {
            #pragma unroll
            for (int r = 0; r < 8; ++r) {
                const int i = mbase + mi * 16 + r + 8 * hi;
                const int j = nbase + ni * 16 + lx;
                // exp(cos/T - 1/T) == exp((cos - 1) * (1/T)); diagonal -> 0
                const float e = __expf((acc[mi][ni][r] - 1.0f) * INV_T);
                esum[r] += (i == j) ? 0.0f : e;
            }
        }
        #pragma unroll
        for (int r = 0; r < 8; ++r) {
            float v = esum[r];
            // reduce across the 16 lanes of each half-wave N-group
            v = SWZ_ADD(v, SWAPX1);
            v = SWZ_ADD(v, SWAPX2);
            v = SWZ_ADD(v, SWAPX4);
            v = SWZ_ADD(v, SWAPX8);
            if (lx == 0)
                atomicAdd(&rowsum[mbase + mi * 16 + r + 8 * hi], v);
        }
    }
}

// ---------------------------------------------------------------------------
// Kernel 3: per-row pair logit (one 1024-wide f16 dot per row, fp32 accum)
// then loss = mean_i( log(rowsum[i]) + 1/T - dot_i/T ).
// One wave per row; 8 rows per 256-thread block.
// ---------------------------------------------------------------------------
__global__ __launch_bounds__(256)
void vcl_finalize(const _Float16* __restrict__ fn, const float* __restrict__ rowsum,
                  float* __restrict__ out) {
    const int lane = threadIdx.x & 31;
    const int wave = threadIdx.x >> 5;
    const int i    = blockIdx.x * 8 + wave;
    const int p    = (i + HALF_N) & (N_ROWS - 1);

    const _Float16* pa = fn + (size_t)i * DIM + lane * 32;
    const _Float16* pb = fn + (size_t)p * DIM + lane * 32;

    float dot = 0.0f;
    #pragma unroll
    for (int c = 0; c < 2; ++c) {
        const v16h xa = ((const v16h*)pa)[c];
        const v16h xb = ((const v16h*)pb)[c];
        #pragma unroll
        for (int e = 0; e < 16; ++e)
            dot = fmaf((float)xa[e], (float)xb[e], dot);
    }
    dot = wave32_sum(dot);

    __shared__ float ws[8];
    if (lane == 0)
        ws[wave] = __logf(rowsum[i]) + (1.0f - dot) * INV_T;
    __syncthreads();
    if (threadIdx.x == 0) {
        float s = 0.f;
        #pragma unroll
        for (int w = 0; w < 8; ++w) s += ws[w];
        atomicAdd(out, s * (1.0f / (float)N_ROWS));
    }
}

// ---------------------------------------------------------------------------
extern "C" void kernel_launch(void* const* d_in, const int* in_sizes, int n_in,
                              void* d_out, int out_size, void* d_ws, size_t ws_size,
                              hipStream_t stream) {
    (void)in_sizes; (void)n_in; (void)out_size; (void)ws_size;
    const float* features = (const float*)d_in[0];
    float* out = (float*)d_out;

    // workspace layout: f16 normalized matrix (16 MB) | rowsum (32 KB)
    _Float16* fn   = (_Float16*)d_ws;
    float*    rsum = (float*)((char*)d_ws + (size_t)N_ROWS * DIM * sizeof(_Float16));

    vcl_normalize<<<N_ROWS, 256, 0, stream>>>(features, fn, rsum, out);

    dim3 grid(N_ROWS / 128, N_ROWS / 128);
    vcl_gemm<<<grid, 256, 0, stream>>>(fn, rsum);

    vcl_finalize<<<N_ROWS / 8, 256, 0, stream>>>(fn, rsum, out);
}
